// ResDetHeaBlk_v1_28321014350464
// MI455X (gfx1250) — matrix operand order repaired and checked by reading, not compiled, not run on hardware
//
#include <hip/hip_runtime.h>

// ---------------------------------------------------------------------------
// Types for CDNA5 WMMA
// ---------------------------------------------------------------------------
typedef __attribute__((ext_vector_type(16))) __bf16 v16bf;
typedef __attribute__((ext_vector_type(8)))  float  v8f;

#define M_PIX   32768          // 8 * 64 * 64
#define COUT    576
#define TS      40             // LDS cell/row K-stride in halfwords (32 data + 8 pad)

__device__ __forceinline__ unsigned short f2bf(float f) {
    unsigned int u = __float_as_uint(f);
    u = u + 0x7FFFu + ((u >> 16) & 1u);           // round-to-nearest-even
    return (unsigned short)(u >> 16);
}
__device__ __forceinline__ float bf2f(unsigned short s) {
    return __uint_as_float(((unsigned int)s) << 16);
}

// ---------------------------------------------------------------------------
// BN fold: scale = g / sqrt(v+eps), bias = b - m*scale
// ---------------------------------------------------------------------------
__global__ void prep_bn_kernel(const float* __restrict__ g, const float* __restrict__ b,
                               const float* __restrict__ m, const float* __restrict__ v,
                               float* __restrict__ scale, float* __restrict__ bias) {
    int c = blockIdx.x * blockDim.x + threadIdx.x;
    if (c >= COUT) return;
    float inv = g[c] * rsqrtf(v[c] + 1e-5f);
    scale[c] = inv;
    bias[c]  = b[c] - m[c] * inv;
}

// ---------------------------------------------------------------------------
// Weight repack: OIHW f32 -> [tap][O][I] bf16   (tap = ky*3+kx)
// ---------------------------------------------------------------------------
__global__ void prep_w_kernel(const float* __restrict__ w, unsigned short* __restrict__ out,
                              int cin) {
    long idx = (long)blockIdx.x * blockDim.x + threadIdx.x;
    long per = (long)COUT * cin;
    long total = 9L * per;
    if (idx >= total) return;
    long t = idx / per;
    long r = idx - t * per;
    long o = r / cin;
    long i = r - o * cin;
    out[idx] = f2bf(w[(o * cin + i) * 9 + t]);
}

// ---------------------------------------------------------------------------
// Build X0 = concat(coarse_m[384], patch_major(raw_x)[192]) -> NHWC bf16 [M,576]
// ---------------------------------------------------------------------------
__global__ void build_x0_kernel(const float* __restrict__ cm, const float* __restrict__ raw,
                                unsigned short* __restrict__ X0) {
    long idx = (long)blockIdx.x * blockDim.x + threadIdx.x;
    const long total = (long)M_PIX * 576;
    if (idx >= total) return;
    int  c   = (int)(idx % 576);
    long pix = idx / 576;
    int b = (int)(pix >> 12), y = (int)((pix >> 6) & 63), x = (int)(pix & 63);
    float v;
    if (c < 384) {
        v = cm[(((long)b * 384 + c) << 12) + (y << 6) + x];
    } else {
        int j  = c - 384;
        int pg = j / 3, cr = j - pg * 3;          // pg = py*8+px
        int pyv = pg >> 3, pxv = pg & 7;
        v = raw[(((long)b * 3 + cr) * 512 + (pyv * 64 + y)) * 512 + (pxv * 64 + x)];
    }
    X0[idx] = f2bf(v);
}

// ---------------------------------------------------------------------------
// NCHW f32 -> NHWC bf16 slice of the 1632-channel concat buffer
// ---------------------------------------------------------------------------
__global__ void nchw2nhwc_kernel(const float* __restrict__ src, unsigned short* __restrict__ dst,
                                 int C, int coff, int cstride) {
    long idx = (long)blockIdx.x * blockDim.x + threadIdx.x;
    long total = (long)M_PIX * C;
    if (idx >= total) return;
    int  c   = (int)(idx % C);
    long pix = idx / C;
    int b = (int)(pix >> 12), y = (int)((pix >> 6) & 63), x = (int)(pix & 63);
    float v = src[(((long)b * C + c) << 12) + (y << 6) + x];
    dst[pix * (long)cstride + coff + c] = f2bf(v);
}

// ---------------------------------------------------------------------------
// 3x3 conv as implicit GEMM with a halo activation tile in LDS.
//   Block tile: 256 pixels (4 image rows x 64 cols, batch-aligned) x 64 out ch.
//   A halo: 6 rows x 66 cols x 32 ch staged ONCE per channel chunk, all 9 taps
//   read shifted windows out of it (9x less global A traffic).
//   B: 64x32 weight tile per (chunk, tap), double buffered.
//   8 waves, each owns 32(M) x 64(N): 8 v_wmma per tap, 72 per chunk.
// ---------------------------------------------------------------------------
template<int CIN>
__global__ __launch_bounds__(256) void conv3x3_wmma_halo(
    const unsigned short* __restrict__ Xin,
    const unsigned short* __restrict__ Wt,
    const float* __restrict__ scale, const float* __restrict__ bias,
    unsigned short* __restrict__ Yout,
    int outStride, int outCoff)
{
    constexpr int NC = CIN / 32;
    __shared__ unsigned short sA[6 * 66 * TS];        // 31680 B halo tile
    __shared__ unsigned short sB[2][64 * TS];         // 2 x 5120 B weight tiles

    const int tid  = threadIdx.x;
    const int wid  = tid >> 5;
    const int lane = tid & 31;
    const int nBase = blockIdx.x * 64;
    const int mBase = blockIdx.y * 256;
    const int bImg  = mBase >> 12;
    const int yBase = (mBase & 4095) >> 6;            // multiple of 4
    const int oct   = tid & 3;

    // ---- A halo staging precompute: 396 cells x 4 octs = 1584 slots, 7/thread ----
    int  aSrc[7]; int aLds[7]; bool aOK[7]; bool aIn[7];
#pragma unroll
    for (int j = 0; j < 7; ++j) {
        int cell = (tid >> 2) + j * 64;
        int hr = cell / 66, hc = cell - hr * 66;
        int sy = yBase + hr - 1, sx = hc - 1;
        aIn[j] = (cell < 396);
        aOK[j] = aIn[j] && ((unsigned)sy < 64u) && ((unsigned)sx < 64u);
        int pix = (bImg << 12) + (sy << 6) + sx;
        aSrc[j] = aOK[j] ? (pix * CIN + oct * 8) : 0;
        aLds[j] = cell * TS + oct * 8;
    }
    // ---- B staging precompute ----
    const int bRow = tid >> 2;
    const int bLds = bRow * TS + oct * 8;

    uint4 rA[7]; uint4 rB;

    auto gloadA = [&](int cc) {
        const int c0 = cc << 5;
#pragma unroll
        for (int j = 0; j < 7; ++j)
            rA[j] = aOK[j] ? *(const uint4*)(Xin + (size_t)(aSrc[j] + c0))
                           : make_uint4(0u, 0u, 0u, 0u);
    };
    auto storeA = [&]() {
#pragma unroll
        for (int j = 0; j < 7; ++j)
            if (aIn[j]) *(uint4*)&sA[aLds[j]] = rA[j];
    };
    auto gloadB = [&](int cc, int t) {
        rB = *(const uint4*)(Wt + (size_t)(t * COUT + nBase + bRow) * CIN + (cc << 5) + oct * 8);
    };
    auto storeB = [&](int buf) { *(uint4*)&sB[buf][bLds] = rB; };

    // ---- fragment position precompute ----
    const int wm = wid * 32;                          // wave's 32 M rows
    int offA[2];
#pragma unroll
    for (int s = 0; s < 2; ++s) {
        int p = wm + s * 16 + (lane & 15);            // 16 consecutive cols, same image row
        int r = p >> 6, c = p & 63;
        offA[s] = ((r + 1) * 66 + (c + 1)) * TS + ((lane >> 4) << 3);
    }
    const int colLane = lane & 15;
    const int kbB = (lane >> 4) << 3;

    float sc4[4], bs4[4];
#pragma unroll
    for (int q = 0; q < 4; ++q) {
        sc4[q] = scale[nBase + q * 16 + colLane];
        bs4[q] = bias [nBase + q * 16 + colLane];
    }

    v8f zero = {};
    v8f acc[2][4];
#pragma unroll
    for (int s = 0; s < 2; ++s)
#pragma unroll
        for (int q = 0; q < 4; ++q) acc[s][q] = zero;

    // ---- main pipeline ----
    gloadA(0);
    gloadB(0, 0);
    int bufB = 0;
    for (int cc = 0; cc < NC; ++cc) {
        __syncthreads();                              // all reads of sA (prev chunk) done
        storeA();
#pragma unroll
        for (int t = 0; t < 9; ++t) {
            storeB(bufB);
            __syncthreads();
            if (t < 8)              gloadB(cc, t + 1);
            else if (cc + 1 < NC) { gloadB(cc + 1, 0); gloadA(cc + 1); }

            const int dy = t / 3 - 1, dx = t - (t / 3) * 3 - 1;   // constants (unrolled)
            const int dOff = (dy * 66 + dx) * TS;

            v16bf a[2];
#pragma unroll
            for (int s = 0; s < 2; ++s) {
                union { uint4 q[2]; v16bf v; } u;
                const unsigned short* pA = sA + offA[s] + dOff;
                u.q[0] = *(const uint4*)(pA);
                u.q[1] = *(const uint4*)(pA + 16);
                a[s] = u.v;
            }
#pragma unroll
            for (int q = 0; q < 4; ++q) {
                union { uint4 w[2]; v16bf v; } u;
                const unsigned short* pB = &sB[bufB][(q * 16 + colLane) * TS + kbB];
                u.w[0] = *(const uint4*)(pB);
                u.w[1] = *(const uint4*)(pB + 16);
                v16bf bq = u.v;
                acc[0][q] = __builtin_amdgcn_wmma_f32_16x16x32_bf16(false, a[0], false, bq, (short)0, acc[0][q], false, false);
                acc[1][q] = __builtin_amdgcn_wmma_f32_16x16x32_bf16(false, a[1], false, bq, (short)0, acc[1][q], false, false);
            }
            bufB ^= 1;
        }
    }

    // ---- epilogue: folded BN, f32 -> bf16, NHWC store ----
    const int mTop = mBase + wm + ((lane >> 4) << 3);
#pragma unroll
    for (int r = 0; r < 8; ++r) {
#pragma unroll
        for (int s = 0; s < 2; ++s) {
            size_t rowOff = (size_t)(mTop + s * 16 + r) * outStride + outCoff;
#pragma unroll
            for (int q = 0; q < 4; ++q)
                Yout[rowOff + nBase + q * 16 + colLane] = f2bf(acc[s][q][r] * sc4[q] + bs4[q]);
        }
    }
}

// ---------------------------------------------------------------------------
// 1x1 conv 576 -> 1 : one wave per pixel, shfl reduction
// ---------------------------------------------------------------------------
__global__ __launch_bounds__(256) void conv1x1_kernel(
    const unsigned short* __restrict__ Y3, const float* __restrict__ w,
    const float* __restrict__ bptr, float* __restrict__ sm)
{
    int wid  = threadIdx.x >> 5;
    int lane = threadIdx.x & 31;
    int pix  = blockIdx.x * 8 + wid;
    if (pix >= M_PIX) return;
    const unsigned short* row = Y3 + (size_t)pix * COUT;
    float s = 0.f;
    for (int c = lane; c < COUT; c += 32)
        s += bf2f(row[c]) * w[c];
#pragma unroll
    for (int off = 16; off; off >>= 1)
        s += __shfl_down(s, off, 32);
    if (lane == 0) sm[pix] = s + bptr[0];
}

// ---------------------------------------------------------------------------
// x8 bilinear upsample, align_corners=True : [8,1,64,64] -> [8,1,512,512] f32
// ---------------------------------------------------------------------------
__global__ void up8_kernel(const float* __restrict__ sm, float* __restrict__ out) {
    int idx = blockIdx.x * blockDim.x + threadIdx.x;
    if (idx >= 8 * 512 * 512) return;
    int b = idx >> 18, rem = idx & 262143;
    int Y = rem >> 9,  X = rem & 511;
    const float sc = 63.0f / 511.0f;
    float ys = Y * sc, xs = X * sc;
    int y0 = (int)ys, x0 = (int)xs;
    int y1 = min(y0 + 1, 63), x1 = min(x0 + 1, 63);
    float wy = ys - (float)y0, wx = xs - (float)x0;
    const float* base = sm + ((size_t)b << 12);
    float v00 = base[(y0 << 6) + x0], v01 = base[(y0 << 6) + x1];
    float v10 = base[(y1 << 6) + x0], v11 = base[(y1 << 6) + x1];
    float r0 = v00 * (1.f - wy) + v10 * wy;
    float r1 = v01 * (1.f - wy) + v11 * wy;
    out[idx] = r0 * (1.f - wx) + r1 * wx;
}

// ---------------------------------------------------------------------------
// Launch
// ---------------------------------------------------------------------------
extern "C" void kernel_launch(void* const* d_in, const int* in_sizes, int n_in,
                              void* d_out, int out_size, void* d_ws, size_t ws_size,
                              hipStream_t stream) {
    const float* coarse_m = (const float*)d_in[0];
    const float* x2_2     = (const float*)d_in[1];
    const float* x3_1     = (const float*)d_in[2];
    const float* raw_x    = (const float*)d_in[3];
    // d_in[4], d_in[5]: w_pred / b_pred -> dead branch, skipped
    const float* dec_w = (const float*)d_in[6];
    const float* dec_g = (const float*)d_in[7];
    const float* dec_b = (const float*)d_in[8];
    const float* dec_m = (const float*)d_in[9];
    const float* dec_v = (const float*)d_in[10];
    const float* cc_w  = (const float*)d_in[11];
    const float* cc_g  = (const float*)d_in[12];
    const float* cc_b  = (const float*)d_in[13];
    const float* cc_m  = (const float*)d_in[14];
    const float* cc_v  = (const float*)d_in[15];
    const float* o1_w  = (const float*)d_in[16];
    const float* o1_g  = (const float*)d_in[17];
    const float* o1_b  = (const float*)d_in[18];
    const float* o1_m  = (const float*)d_in[19];
    const float* o1_v  = (const float*)d_in[20];
    const float* o2_w  = (const float*)d_in[21];
    const float* o2_b  = (const float*)d_in[22];
    float* out = (float*)d_out;

    // Workspace carve-up (256B aligned)
    char* ws = (char*)d_ws;
    size_t off = 0;
    auto alloc = [&](size_t bytes) {
        size_t o = (off + 255) & ~(size_t)255;
        off = o + bytes;
        return o;
    };
    unsigned short* Xcat = (unsigned short*)(ws + alloc((size_t)M_PIX * 1632 * 2));
    unsigned short* X0   = (unsigned short*)(ws + alloc((size_t)M_PIX * 576 * 2));   // reused as Y3
    unsigned short* Y2   = (unsigned short*)(ws + alloc((size_t)M_PIX * 576 * 2));
    unsigned short* Wdec = (unsigned short*)(ws + alloc((size_t)9 * 576 * 576 * 2));
    unsigned short* Wcc  = (unsigned short*)(ws + alloc((size_t)9 * 576 * 1632 * 2));
    unsigned short* Wo1  = (unsigned short*)(ws + alloc((size_t)9 * 576 * 576 * 2));
    float* sbuf = (float*)(ws + alloc((size_t)6 * 576 * 4));
    float* sdec = sbuf + 0 * 576; float* bdec = sbuf + 1 * 576;
    float* scc  = sbuf + 2 * 576; float* bcc  = sbuf + 3 * 576;
    float* so1  = sbuf + 4 * 576; float* bo1  = sbuf + 5 * 576;
    float* smallmap = (float*)(ws + alloc((size_t)M_PIX * 4));

    // --- BN fold ---
    prep_bn_kernel<<<3, 256, 0, stream>>>(dec_g, dec_b, dec_m, dec_v, sdec, bdec);
    prep_bn_kernel<<<3, 256, 0, stream>>>(cc_g,  cc_b,  cc_m,  cc_v,  scc,  bcc);
    prep_bn_kernel<<<3, 256, 0, stream>>>(o1_g,  o1_b,  o1_m,  o1_v,  so1,  bo1);

    // --- weight repack OIHW f32 -> [tap][O][I] bf16 ---
    {
        long t1 = 9L * 576 * 576;
        prep_w_kernel<<<(int)((t1 + 255) / 256), 256, 0, stream>>>(dec_w, Wdec, 576);
        prep_w_kernel<<<(int)((t1 + 255) / 256), 256, 0, stream>>>(o1_w,  Wo1,  576);
        long t2 = 9L * 576 * 1632;
        prep_w_kernel<<<(int)((t2 + 255) / 256), 256, 0, stream>>>(cc_w,  Wcc,  1632);
    }

    // --- build activation buffers ---
    {
        long t0 = (long)M_PIX * 576;
        build_x0_kernel<<<(int)((t0 + 255) / 256), 256, 0, stream>>>(coarse_m, raw_x, X0);
        long ta = (long)M_PIX * 768;
        nchw2nhwc_kernel<<<(int)((ta + 255) / 256), 256, 0, stream>>>(x2_2, Xcat, 768, 0, 1632);
        long tb = (long)M_PIX * 288;
        nchw2nhwc_kernel<<<(int)((tb + 255) / 256), 256, 0, stream>>>(x3_1, Xcat, 288, 1344, 1632);
    }

    // --- the three WMMA convolutions ---
    dim3 grid(COUT / 64, M_PIX / 256);   // (9, 128)
    // dec: X0[.,576] -> Xcat channels 768..1343
    conv3x3_wmma_halo<576><<<grid, 256, 0, stream>>>(X0, Wdec, sdec, bdec, Xcat, 1632, 768);
    // cc: Xcat[.,1632] -> Y2[.,576]
    conv3x3_wmma_halo<1632><<<grid, 256, 0, stream>>>(Xcat, Wcc, scc, bcc, Y2, 576, 0);
    // o1: Y2[.,576] -> X0 (reused as Y3)
    conv3x3_wmma_halo<576><<<grid, 256, 0, stream>>>(Y2, Wo1, so1, bo1, X0, 576, 0);

    // --- 1x1 conv to single channel ---
    conv1x1_kernel<<<M_PIX / 8, 256, 0, stream>>>(X0, o2_w, o2_b, smallmap);

    // --- x8 bilinear upsample (align_corners) ---
    up8_kernel<<<(8 * 512 * 512 + 255) / 256, 256, 0, stream>>>(smallmap, out);
}
